// Attention_77927886618996
// MI455X (gfx1250) — matrix operand order repaired and checked by reading, not compiled, not run on hardware
//
#include <hip/hip_runtime.h>

// ---------------------------------------------------------------------------
// Problem constants (from reference): x (2048,512,8) f32; Q/K/D (8,64,512) f32
// out (2048, 512, 8) f32
// ---------------------------------------------------------------------------
#define NE    2048
#define FEAT  512
#define REST  8
#define HEAD  8
#define LEN   64
#define HL    512          // HEAD*LEN
#define LRDIM 512          // LEN*REST
#define MPROJ 16384        // NE*REST

typedef __attribute__((ext_vector_type(16))) __bf16 v16bf;
typedef __attribute__((ext_vector_type(8)))  __bf16 v8bf;
typedef __attribute__((ext_vector_type(8)))  float  v8f;

// ---------------------------------------------------------------------------
// WMMA fragment loaders (wave32, v_wmma_f32_16x16x32_bf16).
// A is M x K row-major, B is N x K row-major (NT GEMM: C[m,n] = sum_k A[m,k]B[n,k]).
//
// A 16x32 layout (ISA 7.12.2): lanes 0-15 row M=lane hold K {k0..k0+7, k0+16..k0+23};
//                              lanes 16-31 row M=lane-16 hold K {k0+8..15, k0+24..31}.
// B 32x16 layout: lanes 0-15 col N=lane hold K {k0..k0+7, k0+16..k0+23};
//                 lanes 16-31 col N=lane-16 hold K {k0+8..15, k0+24..31}.
// ---------------------------------------------------------------------------
static __device__ __forceinline__ v16bf load_a_frag(const __bf16* __restrict__ A,
                                                    int lda, int row0, int k0) {
    int lane = threadIdx.x & 31;
    int row  = row0 + (lane & 15);
    int kb   = k0 + ((lane >> 4) << 3);          // +8 for hi half-wave
    const __bf16* p = A + (size_t)row * lda + kb;
    v8bf lo = *(const v8bf*)(p);                 // K: kb .. kb+7
    v8bf hi = *(const v8bf*)(p + 16);            // K: kb+16 .. kb+23
    return __builtin_shufflevector(lo, hi, 0,1,2,3,4,5,6,7,8,9,10,11,12,13,14,15);
}

static __device__ __forceinline__ v16bf load_b_frag(const __bf16* __restrict__ B,
                                                    int ldb, int col0, int k0) {
    int lane = threadIdx.x & 31;
    int col  = col0 + (lane & 15);
    int kb   = k0 + ((lane >> 4) << 3);          // +8 for hi half-wave
    const __bf16* p = B + (size_t)col * ldb + kb;
    v8bf lo = *(const v8bf*)(p);
    v8bf hi = *(const v8bf*)(p + 16);
    return __builtin_shufflevector(lo, hi, 0,1,2,3,4,5,6,7,8,9,10,11,12,13,14,15);
}

// One wave computes a 32x32 tile of C with 2x2 WMMAs per K-step (2x frag reuse).
static __device__ __forceinline__ void gemm32x32(const __bf16* __restrict__ A, int lda,
                                                 const __bf16* __restrict__ B, int ldb,
                                                 int tileM, int tileN, int K,
                                                 v8f acc[2][2]) {
    v8f z = {0.f, 0.f, 0.f, 0.f, 0.f, 0.f, 0.f, 0.f};
    acc[0][0] = z; acc[0][1] = z; acc[1][0] = z; acc[1][1] = z;
    int lane = threadIdx.x & 31;
    for (int k0 = 0; k0 < K; k0 += 32) {
        if (k0 + 32 < K) {
            // WGP-scope prefetch of the next K-slab (all 32 rows / cols of tile)
            __builtin_prefetch(A + (size_t)(tileM + lane) * lda + k0 + 32, 0, 3);
            __builtin_prefetch(B + (size_t)(tileN + lane) * ldb + k0 + 32, 0, 3);
        }
        v16bf a0 = load_a_frag(A, lda, tileM,      k0);
        v16bf a1 = load_a_frag(A, lda, tileM + 16, k0);
        v16bf b0 = load_b_frag(B, ldb, tileN,      k0);
        v16bf b1 = load_b_frag(B, ldb, tileN + 16, k0);
        acc[0][0] = __builtin_amdgcn_wmma_f32_16x16x32_bf16(false, a0, false, b0,
                        (short)0, acc[0][0], false, false);
        acc[0][1] = __builtin_amdgcn_wmma_f32_16x16x32_bf16(false, a0, false, b1,
                        (short)0, acc[0][1], false, false);
        acc[1][0] = __builtin_amdgcn_wmma_f32_16x16x32_bf16(false, a1, false, b0,
                        (short)0, acc[1][0], false, false);
        acc[1][1] = __builtin_amdgcn_wmma_f32_16x16x32_bf16(false, a1, false, b1,
                        (short)0, acc[1][1], false, false);
    }
}

// C/D tile layout (ISA 7.12.2): VGPR e holds M=e (lanes 0-15) / M=8+e (lanes 16-31),
// N = lane & 15.
#define EPILOGUE_BEGIN                                                         \
    int lane    = threadIdx.x & 31;                                            \
    int colLane = lane & 15;                                                   \
    int rowHi   = (lane >> 4) << 3;                                            \
    _Pragma("unroll") for (int ti = 0; ti < 2; ++ti)                           \
    _Pragma("unroll") for (int tj = 0; tj < 2; ++tj)                           \
    _Pragma("unroll") for (int e = 0; e < 8; ++e) {                            \
        int row = tileM + ti * 16 + rowHi + e;                                 \
        int col = tileN + tj * 16 + colLane;                                   \
        float v = acc[ti][tj][e];
#define EPILOGUE_END }

// ---------------------------------------------------------------------------
// Stage 0: pack/convert inputs to bf16 in GEMM-friendly layouts
// ---------------------------------------------------------------------------
__global__ void __launch_bounds__(256) convert_x_kernel(const float* __restrict__ x,
                                                        __bf16* __restrict__ Xb) {
    int idx = blockIdx.x * 256 + threadIdx.x;    // MPROJ * FEAT threads
    int f = idx & (FEAT - 1);
    int m = idx >> 9;                            // m = n*8 + r
    int n = m >> 3, r = m & 7;
    Xb[idx] = (__bf16)x[((size_t)n * FEAT + f) * REST + r];
}

__global__ void __launch_bounds__(256) convert_w_kernel(const float* __restrict__ Q,
                                                        const float* __restrict__ K,
                                                        const float* __restrict__ D,
                                                        __bf16* __restrict__ Wb) {
    int idx = blockIdx.x * 256 + threadIdx.x;    // 3 * HL * FEAT threads
    int which = idx >> 18;                       // HL*FEAT = 262144 = 2^18
    int e = idx & (262144 - 1);
    const float* src = (which == 0) ? Q : (which == 1) ? K : D;
    Wb[idx] = (__bf16)src[e];
}

__global__ void __launch_bounds__(256) zero_f32_kernel(float* __restrict__ p, int n) {
    int i = blockIdx.x * 256 + threadIdx.x;
    if (i < n) p[i] = 0.f;
}

__global__ void __launch_bounds__(256) inv_norm_kernel(const float* __restrict__ n2,
                                                       float* __restrict__ inv, int n) {
    int i = blockIdx.x * 256 + threadIdx.x;
    if (i < n) inv[i] = rsqrtf(fmaxf(n2[i], 1e-30f));
}

// ---------------------------------------------------------------------------
// Stage 1: projection GEMM  y[(n,r)][hl] = Xb[(n,r)][f] . W[hl][f]^T
// Writes raw bf16 y into attention layouts + accumulates per-n sum of squares.
// mode 0 (Q/K): dst[h][n][l*8+r]   mode 1 (D): dst[h][l*8+r][n]
// Per-row sum-of-squares is reduced in-register + cross-lane (shfl_xor within
// each 16-lane half-wave) so only one atomic per output row is issued.
// ---------------------------------------------------------------------------
__global__ void __launch_bounds__(256) proj_kernel(const __bf16* __restrict__ Xb,
                                                   const __bf16* __restrict__ W,
                                                   __bf16* __restrict__ dst,
                                                   float* __restrict__ norm2,
                                                   int mode) {
    int wave  = threadIdx.x >> 5;
    int tileM = blockIdx.x * 64  + (wave >> 2) * 32;   // (n,r) rows
    int tileN = blockIdx.y * 128 + (wave & 3) * 32;    // hl cols
    v8f acc[2][2];
    gemm32x32(Xb, FEAT, W, FEAT, tileM, tileN, FEAT, acc);

    EPILOGUE_BEGIN
        int n = row >> 3, r = row & 7;
        int hh = col >> 6, l = col & 63;
        int lr = l * 8 + r;
        if (mode == 0)
            dst[((size_t)hh * NE + n) * LRDIM + lr] = (__bf16)v;
        else
            dst[((size_t)hh * LRDIM + lr) * NE + n] = (__bf16)v;
    EPILOGUE_END

    // norm^2 accumulation: one atomic per row of the 32x32 tile
    #pragma unroll
    for (int ti = 0; ti < 2; ++ti) {
        #pragma unroll
        for (int e = 0; e < 8; ++e) {
            float ss = acc[ti][0][e] * acc[ti][0][e] + acc[ti][1][e] * acc[ti][1][e];
            ss += __shfl_xor(ss, 1, 32);   // reduce over the 16 lanes holding
            ss += __shfl_xor(ss, 2, 32);   // this row's columns (lanes 0-15 and
            ss += __shfl_xor(ss, 4, 32);   // 16-31 hold *different* rows)
            ss += __shfl_xor(ss, 8, 32);
            int row = tileM + ti * 16 + rowHi + e;
            if (colLane == 0) atomicAdd(&norm2[row >> 3], ss);
        }
    }
}

// ---------------------------------------------------------------------------
// Stage 2 (per head): scores S[n][m] = (yq[n][:].yk[m][:]) * invq[n] * invk[m]
// ---------------------------------------------------------------------------
__global__ void __launch_bounds__(256) scores_kernel(const __bf16* __restrict__ yq,
                                                     const __bf16* __restrict__ yk,
                                                     const float* __restrict__ invq,
                                                     const float* __restrict__ invk,
                                                     float* __restrict__ S) {
    int wave  = threadIdx.x >> 5;
    int tileM = blockIdx.x * 64  + (wave >> 2) * 32;   // n
    int tileN = blockIdx.y * 128 + (wave & 3) * 32;    // m
    v8f acc[2][2];
    gemm32x32(yq, LRDIM, yk, LRDIM, tileM, tileN, LRDIM, acc);
    EPILOGUE_BEGIN
        S[(size_t)row * NE + col] = v * invq[row] * invk[col];
    EPILOGUE_END
}

// ---------------------------------------------------------------------------
// Stage 3 (per head): row softmax over m, fold 1/||d_m|| into the weights,
// emit bf16 probabilities wb[n][m] (B operand of the combine GEMM).
// ---------------------------------------------------------------------------
__global__ void __launch_bounds__(256) softmax_kernel(const float* __restrict__ S,
                                                      const float* __restrict__ invd,
                                                      __bf16* __restrict__ wb) {
    __shared__ float red[256];
    int n = blockIdx.x, tid = threadIdx.x;
    float vals[8];
    float mx = -3.0e38f;
    #pragma unroll
    for (int i = 0; i < 8; ++i) {
        float v = S[(size_t)n * NE + tid + i * 256];
        vals[i] = v;
        mx = fmaxf(mx, v);
    }
    red[tid] = mx; __syncthreads();
    for (int s = 128; s > 0; s >>= 1) {
        if (tid < s) red[tid] = fmaxf(red[tid], red[tid + s]);
        __syncthreads();
    }
    mx = red[0]; __syncthreads();
    float sum = 0.f;
    #pragma unroll
    for (int i = 0; i < 8; ++i) {
        float e = __expf(vals[i] - mx);
        vals[i] = e;
        sum += e;
    }
    red[tid] = sum; __syncthreads();
    for (int s = 128; s > 0; s >>= 1) {
        if (tid < s) red[tid] += red[tid + s];
        __syncthreads();
    }
    float inv = 1.0f / red[0];
    #pragma unroll
    for (int i = 0; i < 8; ++i) {
        int m = tid + i * 256;
        wb[(size_t)n * NE + m] = (__bf16)(vals[i] * inv * invd[m]);
    }
}

// ---------------------------------------------------------------------------
// Stage 4 (per head): vT[lr][n] = sum_m yd[lr][m] * wb[n][m]
// scatter into out[n][h*64+l][r] = out[n*4096 + h*512 + lr]
// ---------------------------------------------------------------------------
__global__ void __launch_bounds__(256) combine_kernel(const __bf16* __restrict__ yd,
                                                      const __bf16* __restrict__ wb,
                                                      float* __restrict__ out,
                                                      int h) {
    int wave  = threadIdx.x >> 5;
    int tileM = blockIdx.x * 64  + (wave >> 2) * 32;   // lr
    int tileN = blockIdx.y * 128 + (wave & 3) * 32;    // n
    v8f acc[2][2];
    gemm32x32(yd, NE, wb, NE, tileM, tileN, NE, acc);
    EPILOGUE_BEGIN
        out[(size_t)col * (HL * REST) + h * LRDIM + row] = v;
    EPILOGUE_END
}

// ---------------------------------------------------------------------------
// Host launcher
// ---------------------------------------------------------------------------
extern "C" void kernel_launch(void* const* d_in, const int* in_sizes, int n_in,
                              void* d_out, int out_size, void* d_ws, size_t ws_size,
                              hipStream_t stream) {
    (void)in_sizes; (void)n_in; (void)out_size; (void)ws_size;
    const float* x = (const float*)d_in[0];
    const float* Q = (const float*)d_in[1];
    const float* K = (const float*)d_in[2];
    const float* D = (const float*)d_in[3];
    float* out = (float*)d_out;

    char* ws = (char*)d_ws;
    size_t off = 0;
    auto alloc = [&](size_t bytes) -> void* {
        void* p = ws + off;
        off += (bytes + 255) & ~(size_t)255;
        return p;
    };
    __bf16* Xb    = (__bf16*)alloc((size_t)MPROJ * FEAT * 2);       // 16 MB
    __bf16* Wb    = (__bf16*)alloc((size_t)3 * HL * FEAT * 2);      // 1.5 MB
    __bf16* yqb   = (__bf16*)alloc((size_t)HEAD * NE * LRDIM * 2);  // 16 MB
    __bf16* ykb   = (__bf16*)alloc((size_t)HEAD * NE * LRDIM * 2);  // 16 MB
    __bf16* ydb   = (__bf16*)alloc((size_t)HEAD * LRDIM * NE * 2);  // 16 MB
    float*  norm2 = (float*) alloc((size_t)3 * NE * 4);
    float*  invn  = (float*) alloc((size_t)3 * NE * 4);
    float*  S     = (float*) alloc((size_t)NE * NE * 4);            // 16 MB, L2-resident
    __bf16* wb    = (__bf16*)alloc((size_t)NE * NE * 2);            // 8 MB,  L2-resident

    // Stage 0: packing
    zero_f32_kernel<<<(3 * NE + 255) / 256, 256, 0, stream>>>(norm2, 3 * NE);
    convert_x_kernel<<<(MPROJ * FEAT) / 256, 256, 0, stream>>>(x, Xb);
    convert_w_kernel<<<(3 * HL * FEAT) / 256, 256, 0, stream>>>(Q, K, D, Wb);

    // Stage 1: three projection GEMMs (16384 x 512 x 512)
    dim3 gProj(MPROJ / 64, HL / 128);
    proj_kernel<<<gProj, 256, 0, stream>>>(Xb, Wb + 0 * HL * FEAT, yqb, norm2 + 0 * NE, 0);
    proj_kernel<<<gProj, 256, 0, stream>>>(Xb, Wb + 1 * HL * FEAT, ykb, norm2 + 1 * NE, 0);
    proj_kernel<<<gProj, 256, 0, stream>>>(Xb, Wb + 2 * HL * FEAT, ydb, norm2 + 2 * NE, 1);
    inv_norm_kernel<<<(3 * NE + 255) / 256, 256, 0, stream>>>(norm2, invn, 3 * NE);

    // Stages 2-4 per head; S/wb reused, kept hot in the 192MB L2
    dim3 gScore(NE / 64, NE / 128);
    dim3 gComb(LRDIM / 64, NE / 128);
    for (int h = 0; h < HEAD; ++h) {
        scores_kernel<<<gScore, 256, 0, stream>>>(
            yqb + (size_t)h * NE * LRDIM, ykb + (size_t)h * NE * LRDIM,
            invn + 0 * NE, invn + 1 * NE, S);
        softmax_kernel<<<NE, 256, 0, stream>>>(S, invn + 2 * NE, wb);
        combine_kernel<<<gComb, 256, 0, stream>>>(
            ydb + (size_t)h * LRDIM * NE, wb, out, h);
    }
}
